// BasicTransformerBlock_12833362280563
// MI455X (gfx1250) — matrix operand order repaired and checked by reading, not compiled, not run on hardware
//
#include <hip/hip_runtime.h>

typedef __bf16 bf16;
typedef __attribute__((ext_vector_type(16))) __bf16 v16bf;
typedef __attribute__((ext_vector_type(8)))  float  v8f;
typedef __attribute__((ext_vector_type(4)))  unsigned int v4u;
typedef __attribute__((ext_vector_type(8)))  int v8i;
typedef __attribute__((ext_vector_type(4)))  int v4i;

#define LN_EPS 1e-5f

// ---------------- f32 -> bf16 conversion ----------------
__global__ void btb_cvt_kernel(const float* __restrict__ in, bf16* __restrict__ out, int n) {
  int i = blockIdx.x * blockDim.x + threadIdx.x;
  if (i < n) out[i] = (bf16)in[i];
}

// ---------------- LayerNorm over width-1024 rows, f32 in -> bf16 out ----------------
__global__ void btb_ln_kernel(const float* __restrict__ x, const float* __restrict__ w,
                              const float* __restrict__ b, bf16* __restrict__ out) {
  const int C = 1024;
  int row = blockIdx.x;
  int tid = threadIdx.x;  // 256 threads, 4 elems each
  const float* xr = x + (size_t)row * C;
  float v[4];
  float s = 0.f;
#pragma unroll
  for (int j = 0; j < 4; ++j) { v[j] = xr[tid + 256 * j]; s += v[j]; }
  __shared__ float red[256];
  red[tid] = s; __syncthreads();
  for (int off = 128; off > 0; off >>= 1) {
    if (tid < off) red[tid] += red[tid + off];
    __syncthreads();
  }
  float mu = red[0] * (1.f / C);
  __syncthreads();
  float sq = 0.f;
#pragma unroll
  for (int j = 0; j < 4; ++j) { float d = v[j] - mu; sq += d * d; }
  red[tid] = sq; __syncthreads();
  for (int off = 128; off > 0; off >>= 1) {
    if (tid < off) red[tid] += red[tid + off];
    __syncthreads();
  }
  float rstd = rsqrtf(red[0] * (1.f / C) + LN_EPS);
  bf16* orow = out + (size_t)row * C;
#pragma unroll
  for (int j = 0; j < 4; ++j) {
    int c = tid + 256 * j;
    orow[c] = (bf16)((v[j] - mu) * rstd * w[c] + b[c]);
  }
}

// ---------------- Tensor Data Mover: 2-D tile (bf16) global -> LDS ----------------
// D# packing per CDNA5 ISA (2-byte elements, no pad,
// no multicast, no iterate). Groups 2/3 zero (2-D tile: tile_dim3/4 = 0 = unused).
// This toolchain exposes the 6-arg builtin form (extra int32x8 group, zeroed).
__device__ inline void btb_tdm_load_2d(void* lds_dst, const void* gsrc,
                                       unsigned tile_w, unsigned tile_h,
                                       unsigned tensor_w, unsigned tensor_h,
                                       unsigned long long row_stride_elems) {
  unsigned long long ga = (unsigned long long)(size_t)gsrc;
  unsigned lds_off = (unsigned)(size_t)lds_dst;   // low 32 bits = LDS offset
  v4u g0;
  g0.x = 1u;                                                   // count=1 (valid, user mode)
  g0.y = lds_off;                                              // lds_addr
  g0.z = (unsigned)ga;                                         // global_addr[31:0]
  g0.w = (unsigned)((ga >> 32) & 0x01ffffffu) | (2u << 30);    // global_addr[56:32] | type=2
  v8i g1;
  g1[0] = 0x00010000;                                          // data_size=1 (2 bytes)
  g1[1] = (int)((tensor_w & 0xffffu) << 16);                   // tensor_dim0[15:0]
  g1[2] = (int)(((tensor_w >> 16) & 0xffffu) | ((tensor_h & 0xffffu) << 16));
  g1[3] = (int)(((tensor_h >> 16) & 0xffffu) | ((tile_w & 0xffffu) << 16));
  g1[4] = (int)(tile_h & 0xffffu);                             // tile_dim1 (tile_dim2=0)
  g1[5] = (int)(row_stride_elems & 0xffffffffu);               // tensor_dim0_stride[31:0]
  g1[6] = (int)((row_stride_elems >> 32) & 0xffffu);           // stride[47:32]
  g1[7] = 0;
  v4i gz4 = {0, 0, 0, 0};
  v8i gz8 = {0, 0, 0, 0, 0, 0, 0, 0};
  __builtin_amdgcn_tensor_load_to_lds(g0, g1, gz4, gz4, gz8, 0);
}

// ---------------- WMMA fragment helpers ----------------
// A frag from LDS tile sA[128][32] (row-major): lane l = row l&15, kb = 8*(l>>4)
__device__ inline v16bf btb_afrag_lds(const bf16* __restrict__ sA, int mrow, int lane) {
  const bf16* p = sA + (size_t)mrow * 32 + ((lane >> 4) << 3);
  v16bf a;
#pragma unroll
  for (int e = 0; e < 8; ++e) { a[e] = p[e]; a[e + 8] = p[16 + e]; }
  return a;
}
// B frag from transposed LDS panel sB[n][32]: kb = 8*(l>>4), k = kb+e and kb+16+e
__device__ inline v16bf btb_bfrag_lds(const bf16* __restrict__ sB, int ncol, int lane) {
  const bf16* p = sB + (size_t)(ncol + (lane & 15)) * 32 + ((lane >> 4) << 3);
  v16bf b;
#pragma unroll
  for (int e = 0; e < 8; ++e) { b[e] = p[e]; b[e + 8] = p[16 + e]; }
  return b;
}

// ---------------- GEMM: block tile 128x64, 8 waves, LDS-staged, TDM for A --------
// EPI 0: store bf16 row-major; EPI 1: f32 out = residual + acc + bias;
// EPI 2: store bf16 TRANSPOSED (outB[col * M + row]) for attention V consumption.
template <int EPI>
__global__ void btb_gemm_kernel(const bf16* __restrict__ A, const bf16* __restrict__ B,
                                const float* __restrict__ bias, const float* __restrict__ res,
                                bf16* __restrict__ outB, float* __restrict__ outF,
                                int M, int N, int K) {
  __shared__ bf16 sA[128 * 32];  // A tile, row-major [m][k]
  __shared__ bf16 sB[64 * 32];   // B panel, transposed [n][k]
  int tid = threadIdx.x;
  int lane = tid & 31, wave = tid >> 5;
  int half = lane >> 4, l15 = lane & 15;
  int mBlk = blockIdx.y * 128;
  int nBlk = blockIdx.x * 64;
  int stg_k = tid >> 3;          // 0..31
  int stg_n = (tid & 7) * 8;     // 0..56

  v8f acc[4] = {v8f{}, v8f{}, v8f{}, v8f{}};

  for (int k0 = 0; k0 < K; k0 += 32) {
    // --- TDM: async DMA of A tile [128 x 32] into LDS (wave 0 issues; EXEC ignored) ---
    if (wave == 0) {
      btb_tdm_load_2d(sA, A + (size_t)mBlk * K + k0, 32u, 128u,
                      (unsigned)K, (unsigned)M, (unsigned long long)K);
      __builtin_amdgcn_s_wait_tensorcnt(0);
    }
    // --- cooperative transposed staging of B panel [32 x 64] -> sB[n][k] ---
    {
      const bf16* gp = B + (size_t)(k0 + stg_k) * N + nBlk + stg_n;
      uint4 u = *(const uint4*)gp;               // 8 contiguous bf16, coalesced
      const bf16* t8 = (const bf16*)&u;
#pragma unroll
      for (int j = 0; j < 8; ++j) sB[(stg_n + j) * 32 + stg_k] = t8[j];
      if (k0 + 32 < K)                           // prefetch next K-panel of B
        __builtin_prefetch(B + (size_t)(k0 + 32 + stg_k) * N + nBlk + stg_n, 0, 0);
    }
    __syncthreads();

    // preload all fragments into distinct registers, then burst the WMMAs
    v16bf a = btb_afrag_lds(sA, wave * 16 + l15, lane);
    v16bf bf[4];
#pragma unroll
    for (int t = 0; t < 4; ++t) bf[t] = btb_bfrag_lds(sB, t * 16, lane);
#pragma unroll
    for (int t = 0; t < 4; ++t)
      acc[t] = __builtin_amdgcn_wmma_f32_16x16x32_bf16(false, a, false, bf[t], (short)0, acc[t], false, false);
    __syncthreads();
  }

  int rbase = mBlk + wave * 16 + half * 8;
#pragma unroll
  for (int t = 0; t < 4; ++t) {
    int col = nBlk + t * 16 + l15;
#pragma unroll
    for (int r = 0; r < 8; ++r) {
      if (EPI == 0) {
        outB[(size_t)(rbase + r) * N + col] = (bf16)acc[t][r];
      } else if (EPI == 1) {
        size_t idx = (size_t)(rbase + r) * N + col;
        outF[idx] = res[idx] + acc[t][r] + bias[col];
      } else {  // EPI == 2: transposed store Vt[col][row]
        outB[(size_t)col * M + rbase + r] = (bf16)acc[t][r];
      }
    }
  }
}

// ---------------- GEGLU GEMM: 128x64 tile of h1 and h2, fused gelu product ----------
__global__ void btb_geglu_kernel(const bf16* __restrict__ A, const bf16* __restrict__ B,
                                 const float* __restrict__ bias, bf16* __restrict__ out,
                                 int M, int K) {
  const int N = 4096, LDB = 8192;
  __shared__ bf16 sA[128 * 32];
  __shared__ bf16 sB1[64 * 32];
  __shared__ bf16 sB2[64 * 32];
  int tid = threadIdx.x;
  int lane = tid & 31, wave = tid >> 5;
  int half = lane >> 4, l15 = lane & 15;
  int mBlk = blockIdx.y * 128;
  int nBlk = blockIdx.x * 64;
  int stg_k = tid >> 3;
  int stg_n = (tid & 7) * 8;

  v8f acc1[4] = {v8f{}, v8f{}, v8f{}, v8f{}};
  v8f acc2[4] = {v8f{}, v8f{}, v8f{}, v8f{}};

  for (int k0 = 0; k0 < K; k0 += 32) {
    if (wave == 0) {
      btb_tdm_load_2d(sA, A + (size_t)mBlk * K + k0, 32u, 128u,
                      (unsigned)K, (unsigned)M, (unsigned long long)K);
      __builtin_amdgcn_s_wait_tensorcnt(0);
    }
    {
      const bf16* g1p = B + (size_t)(k0 + stg_k) * LDB + nBlk + stg_n;
      const bf16* g2p = g1p + N;
      uint4 u1 = *(const uint4*)g1p;
      uint4 u2 = *(const uint4*)g2p;
      const bf16* t1 = (const bf16*)&u1;
      const bf16* t2 = (const bf16*)&u2;
#pragma unroll
      for (int j = 0; j < 8; ++j) {
        sB1[(stg_n + j) * 32 + stg_k] = t1[j];
        sB2[(stg_n + j) * 32 + stg_k] = t2[j];
      }
      if (k0 + 32 < K) {
        __builtin_prefetch(B + (size_t)(k0 + 32 + stg_k) * LDB + nBlk + stg_n, 0, 0);
        __builtin_prefetch(B + (size_t)(k0 + 32 + stg_k) * LDB + nBlk + N + stg_n, 0, 0);
      }
    }
    __syncthreads();

    v16bf a = btb_afrag_lds(sA, wave * 16 + l15, lane);
    v16bf bf1[4], bf2[4];
#pragma unroll
    for (int t = 0; t < 4; ++t) {
      bf1[t] = btb_bfrag_lds(sB1, t * 16, lane);
      bf2[t] = btb_bfrag_lds(sB2, t * 16, lane);
    }
#pragma unroll
    for (int t = 0; t < 4; ++t) {
      acc1[t] = __builtin_amdgcn_wmma_f32_16x16x32_bf16(false, a, false, bf1[t], (short)0, acc1[t], false, false);
      acc2[t] = __builtin_amdgcn_wmma_f32_16x16x32_bf16(false, a, false, bf2[t], (short)0, acc2[t], false, false);
    }
    __syncthreads();
  }

  int rbase = mBlk + wave * 16 + half * 8;
#pragma unroll
  for (int t = 0; t < 4; ++t) {
    int col = nBlk + t * 16 + l15;
#pragma unroll
    for (int r = 0; r < 8; ++r) {
      float h1 = acc1[t][r] + bias[col];
      float h2 = acc2[t][r] + bias[col + N];
      float g  = h1 * 0.5f * h2 * (1.f + erff(h2 * 0.70710678118f));
      out[(size_t)(rbase + r) * N + col] = (bf16)g;
    }
  }
}

// ---------------- Flash attention: one wave32 per 16-query tile, H=16, D=64 ----------------
// Q, Kv row-major [rows, H*D]; Vt TRANSPOSED [H*D, Mv] (Mv = total kv rows = B*S).
__global__ void btb_attn_kernel(const bf16* __restrict__ Q, const bf16* __restrict__ Kv,
                                const bf16* __restrict__ Vt, bf16* __restrict__ O,
                                int T, int S, int Mv) {
  const int H = 16, D = 64, HD = 1024;
  int lane = threadIdx.x & 31;
  int half = lane >> 4;
  int l15  = lane & 15;
  int q0 = blockIdx.x * 16;
  int h  = blockIdx.y;
  int b  = blockIdx.z;

  const bf16* qbase = Q + ((size_t)(b * T + q0 + l15) * H + h) * D;
  int kb = half << 3;
  v16bf aq0, aq1;
#pragma unroll
  for (int e = 0; e < 8; ++e) {
    aq0[e] = qbase[kb + e];       aq0[e + 8] = qbase[kb + 16 + e];
    aq1[e] = qbase[32 + kb + e];  aq1[e + 8] = qbase[32 + kb + 16 + e];
  }

  v8f acc[4] = {v8f{}, v8f{}, v8f{}, v8f{}};
  float mrow[8], lrow[8];
#pragma unroll
  for (int r = 0; r < 8; ++r) { mrow[r] = -1e30f; lrow[r] = 0.f; }

  __shared__ bf16 pls[16 * 32];

  const bf16* kbase0 = Kv + ((size_t)b * S * H + h) * D;
  const bf16* vbase0 = Vt + (size_t)h * D * Mv + (size_t)b * S;  // Vt[h*64+d][b*S+s]

  for (int j0 = 0; j0 < S; j0 += 32) {
    int koff = half << 3;
    const bf16* krA = kbase0 + (size_t)(j0 + l15) * HD;
    const bf16* krB = kbase0 + (size_t)(j0 + 16 + l15) * HD;
    v16bf bk0, bk1, bk2, bk3;
#pragma unroll
    for (int e = 0; e < 8; ++e) {
      bk0[e] = krA[koff + e];      bk1[e] = krA[32 + koff + e];
      bk2[e] = krB[koff + e];      bk3[e] = krB[32 + koff + e];
      bk0[e + 8] = krA[koff + 16 + e];      bk1[e + 8] = krA[32 + koff + 16 + e];
      bk2[e + 8] = krB[koff + 16 + e];      bk3[e + 8] = krB[32 + koff + 16 + e];
    }
    v8f sA = {}, sB = {};
    sA = __builtin_amdgcn_wmma_f32_16x16x32_bf16(false, aq0, false, bk0, (short)0, sA, false, false);
    sA = __builtin_amdgcn_wmma_f32_16x16x32_bf16(false, aq1, false, bk1, (short)0, sA, false, false);
    sB = __builtin_amdgcn_wmma_f32_16x16x32_bf16(false, aq0, false, bk2, (short)0, sB, false, false);
    sB = __builtin_amdgcn_wmma_f32_16x16x32_bf16(false, aq1, false, bk3, (short)0, sB, false, false);

    float pA[8], pB[8];
#pragma unroll
    for (int r = 0; r < 8; ++r) {
      float sa = sA[r] * 0.125f;
      float sb = sB[r] * 0.125f;
      float mx = fmaxf(sa, sb);
      mx = fmaxf(mx, __shfl_xor(mx, 1, 32));
      mx = fmaxf(mx, __shfl_xor(mx, 2, 32));
      mx = fmaxf(mx, __shfl_xor(mx, 4, 32));
      mx = fmaxf(mx, __shfl_xor(mx, 8, 32));
      float nm = fmaxf(mrow[r], mx);
      pA[r] = __expf(sa - nm);
      pB[r] = __expf(sb - nm);
      float ps = pA[r] + pB[r];
      ps += __shfl_xor(ps, 1, 32);
      ps += __shfl_xor(ps, 2, 32);
      ps += __shfl_xor(ps, 4, 32);
      ps += __shfl_xor(ps, 8, 32);
      float f = __expf(mrow[r] - nm);
      lrow[r] = lrow[r] * f + ps;
      mrow[r] = nm;
#pragma unroll
      for (int t = 0; t < 4; ++t) acc[t][r] *= f;
    }

#pragma unroll
    for (int r = 0; r < 8; ++r) {
      int row = r + 8 * half;
      pls[row * 32 + l15]      = (bf16)pA[r];
      pls[row * 32 + 16 + l15] = (bf16)pB[r];
    }
    __syncthreads();
    v16bf ap;
    {
      const bf16* pr = &pls[l15 * 32 + kb];
#pragma unroll
      for (int e = 0; e < 8; ++e) { ap[e] = pr[e]; ap[e + 8] = pr[16 + e]; }
    }
    __syncthreads();

    // O += P @ V ; V fragments are two 8-bf16 runs (k = koff+e, koff+16+e) from transposed Vt
    v16bf bv[4];
#pragma unroll
    for (int t = 0; t < 4; ++t) {
      const bf16* vr = vbase0 + (size_t)(t * 16 + l15) * Mv + j0 + koff;
#pragma unroll
      for (int e = 0; e < 8; ++e) { bv[t][e] = vr[e]; bv[t][e + 8] = vr[16 + e]; }
    }
#pragma unroll
    for (int t = 0; t < 4; ++t)
      acc[t] = __builtin_amdgcn_wmma_f32_16x16x32_bf16(false, ap, false, bv[t], (short)0, acc[t], false, false);
  }

  bf16* ob = O + ((size_t)(b * T + q0) * H + h) * D;
#pragma unroll
  for (int r = 0; r < 8; ++r) {
    float inv = 1.f / lrow[r];
    int row = r + 8 * half;
#pragma unroll
    for (int t = 0; t < 4; ++t)
      ob[(size_t)row * HD + t * 16 + l15] = (bf16)(acc[t][r] * inv);
  }
}

// ---------------- host orchestration ----------------
extern "C" void kernel_launch(void* const* d_in, const int* in_sizes, int n_in,
                              void* d_out, int out_size, void* d_ws, size_t ws_size,
                              hipStream_t stream) {
  (void)in_sizes; (void)n_in; (void)out_size; (void)ws_size;
  const int B = 2, T = 2048, C = 1024, S = 512, DFF = 4096;
  const int M = B * T;

  const float* x     = (const float*)d_in[0];
  const float* enc   = (const float*)d_in[1];
  const float* ln1w  = (const float*)d_in[2];
  const float* ln1b  = (const float*)d_in[3];
  const float* a1wq  = (const float*)d_in[4];
  const float* a1wk  = (const float*)d_in[5];
  const float* a1wv  = (const float*)d_in[6];
  const float* a1wo  = (const float*)d_in[7];
  const float* a1bo  = (const float*)d_in[8];
  const float* ln2w  = (const float*)d_in[9];
  const float* ln2b  = (const float*)d_in[10];
  const float* a2wq  = (const float*)d_in[11];
  const float* a2wk  = (const float*)d_in[12];
  const float* a2wv  = (const float*)d_in[13];
  const float* a2wo  = (const float*)d_in[14];
  const float* a2bo  = (const float*)d_in[15];
  const float* ln3w  = (const float*)d_in[16];
  const float* ln3b  = (const float*)d_in[17];
  const float* ffw1  = (const float*)d_in[18];
  const float* ffb1  = (const float*)d_in[19];
  const float* ffw2  = (const float*)d_in[20];
  const float* ffb2  = (const float*)d_in[21];

  char* ws = (char*)d_ws;
  size_t off = 0;
  auto take = [&](size_t bytes) -> void* {
    void* p = ws + off;
    off += (bytes + 255) & ~(size_t)255;
    return p;
  };
  bf16* h_bf   = (bf16*)take((size_t)M * C * 2);
  bf16* q_bf   = (bf16*)take((size_t)M * C * 2);
  bf16* k_bf   = (bf16*)take((size_t)M * C * 2);
  bf16* vt_bf  = (bf16*)take((size_t)M * C * 2);   // transposed V: [C][rows]
  bf16* ao_bf  = (bf16*)take((size_t)M * C * 2);
  float* x1    = (float*)take((size_t)M * C * 4);
  float* x2    = (float*)take((size_t)M * C * 4);
  bf16* g_bf   = (bf16*)take((size_t)M * DFF * 2);
  bf16* enc_bf = (bf16*)take((size_t)B * S * 64 * 2);
  bf16* wq1b   = (bf16*)take((size_t)C * C * 2);
  bf16* wk1b   = (bf16*)take((size_t)C * C * 2);
  bf16* wv1b   = (bf16*)take((size_t)C * C * 2);
  bf16* wo1b   = (bf16*)take((size_t)C * C * 2);
  bf16* wq2b   = (bf16*)take((size_t)C * C * 2);
  bf16* wk2b   = (bf16*)take((size_t)64 * C * 2);
  bf16* wv2b   = (bf16*)take((size_t)64 * C * 2);
  bf16* wo2b   = (bf16*)take((size_t)C * C * 2);
  bf16* w1b    = (bf16*)take((size_t)C * 2 * DFF * 2);
  bf16* w2b    = (bf16*)take((size_t)DFF * C * 2);

  auto cvt = [&](const float* src, bf16* dst, int n) {
    btb_cvt_kernel<<<(n + 255) / 256, 256, 0, stream>>>(src, dst, n);
  };
  cvt(a1wq, wq1b, C * C);  cvt(a1wk, wk1b, C * C);
  cvt(a1wv, wv1b, C * C);  cvt(a1wo, wo1b, C * C);
  cvt(a2wq, wq2b, C * C);  cvt(a2wk, wk2b, 64 * C);
  cvt(a2wv, wv2b, 64 * C); cvt(a2wo, wo2b, C * C);
  cvt(ffw1, w1b, C * 2 * DFF);
  cvt(ffw2, w2b, DFF * C);
  cvt(enc, enc_bf, B * S * 64);

  dim3 gMC(C / 64, M / 128);        // 64-wide col blocks, 128-row blocks
  dim3 gKV(C / 64, (B * S) / 128);
  dim3 gAT(T / 16, 16, B);

  // ---- self-attention block ----
  btb_ln_kernel<<<M, 256, 0, stream>>>(x, ln1w, ln1b, h_bf);
  btb_gemm_kernel<0><<<gMC, 256, 0, stream>>>(h_bf, wq1b, nullptr, nullptr, q_bf, nullptr, M, C, C);
  btb_gemm_kernel<0><<<gMC, 256, 0, stream>>>(h_bf, wk1b, nullptr, nullptr, k_bf, nullptr, M, C, C);
  btb_gemm_kernel<2><<<gMC, 256, 0, stream>>>(h_bf, wv1b, nullptr, nullptr, vt_bf, nullptr, M, C, C);
  btb_attn_kernel<<<gAT, 32, 0, stream>>>(q_bf, k_bf, vt_bf, ao_bf, T, T, M);
  btb_gemm_kernel<1><<<gMC, 256, 0, stream>>>(ao_bf, wo1b, a1bo, x, nullptr, x1, M, C, C);

  // ---- cross-attention block ----
  btb_ln_kernel<<<M, 256, 0, stream>>>(x1, ln2w, ln2b, h_bf);
  btb_gemm_kernel<0><<<gMC, 256, 0, stream>>>(h_bf, wq2b, nullptr, nullptr, q_bf, nullptr, M, C, C);
  btb_gemm_kernel<0><<<gKV, 256, 0, stream>>>(enc_bf, wk2b, nullptr, nullptr, k_bf, nullptr, B * S, C, 64);
  btb_gemm_kernel<2><<<gKV, 256, 0, stream>>>(enc_bf, wv2b, nullptr, nullptr, vt_bf, nullptr, B * S, C, 64);
  btb_attn_kernel<<<gAT, 32, 0, stream>>>(q_bf, k_bf, vt_bf, ao_bf, T, S, B * S);
  btb_gemm_kernel<1><<<gMC, 256, 0, stream>>>(ao_bf, wo2b, a2bo, x1, nullptr, x2, M, C, C);

  // ---- GEGLU feed-forward block ----
  btb_ln_kernel<<<M, 256, 0, stream>>>(x2, ln3w, ln3b, h_bf);
  btb_geglu_kernel<<<dim3(DFF / 64, M / 128), 256, 0, stream>>>(h_bf, w1b, ffb1, g_bf, M, C);
  btb_gemm_kernel<1><<<gMC, 256, 0, stream>>>(g_bf, w2b, ffb2, x2, nullptr, (float*)d_out, M, C, DFF);
}
